// VGNConvLayer_v2_34041910788488
// MI455X (gfx1250) — matrix operand order repaired and checked by reading, not compiled, not run on hardware
//
#include <hip/hip_runtime.h>

#define NN 100000
#define NE 1600000
#define DIM 128
#define LAYERS 3
#define BN_EPS 1e-5f

typedef __attribute__((ext_vector_type(16))) __bf16 v16bf;
typedef __attribute__((ext_vector_type(8)))  float  v8f;

union Frag {
    v16bf v;
    uint4 q[2];
};

static __device__ __forceinline__ unsigned short f2bf(float f) {
    unsigned int u = __float_as_uint(f);
    unsigned int r = (u + 0x7FFFu + ((u >> 16) & 1u)) >> 16;   // round-to-nearest-even
    return (unsigned short)r;
}

// ---------------- weight prep: Wt_bf16[l][which][j][k] = bf16(W[l][k][j]) ----
__global__ void k_wprep(const float* __restrict__ W1, const float* __restrict__ W2,
                        unsigned short* __restrict__ wt) {
    int t = blockIdx.x * blockDim.x + threadIdx.x;
    if (t >= LAYERS * 2 * DIM * DIM) return;
    int k  = t & 127;
    int j  = (t >> 7) & 127;
    int lw = t >> 14;                   // l*2 + which
    int l  = lw >> 1, which = lw & 1;
    const float* W = which ? W2 : W1;
    wt[t] = f2bf(W[(size_t)l * DIM * DIM + (size_t)k * DIM + j]);
}

// ---------------- scatter: one wave per edge, atomic f32 add into agg --------
__global__ void k_scatter(const float* __restrict__ h, const float* __restrict__ ea,
                          const int* __restrict__ ei, float* __restrict__ agg) {
    int gid  = blockIdx.x * blockDim.x + threadIdx.x;
    int e    = gid >> 5;
    int lane = gid & 31;
    if (e >= NE) return;
    int s = ei[e];
    int d = ei[NE + e];
    int c = lane * 4;
    float4 hv = *(const float4*)(h  + (size_t)s * DIM + c);
    float4 av = *(const float4*)(ea + (size_t)e * DIM + c);
    float4 m;
    m.x = fmaxf(hv.x + av.x, 0.f);
    m.y = fmaxf(hv.y + av.y, 0.f);
    m.z = fmaxf(hv.z + av.z, 0.f);
    m.w = fmaxf(hv.w + av.w, 0.f);
    float* ap = agg + (size_t)d * DIM + c;
    atomicAdd(ap + 0, m.x);
    atomicAdd(ap + 1, m.y);
    atomicAdd(ap + 2, m.z);
    atomicAdd(ap + 3, m.w);
}

// ---------------- z = (1+eps)*h + agg, cast to bf16 --------------------------
__global__ void k_zprep(const float* __restrict__ h, const float* __restrict__ agg,
                        const float* __restrict__ eps, int l,
                        unsigned short* __restrict__ zbf) {
    int i = (blockIdx.x * blockDim.x + threadIdx.x) * 4;
    if (i >= NN * DIM) return;
    float e1 = 1.0f + eps[l];
    float4 hv = *(const float4*)(h + i);
    float4 av = *(const float4*)(agg + i);
    ushort4 o;
    o.x = f2bf(e1 * hv.x + av.x);
    o.y = f2bf(e1 * hv.y + av.y);
    o.z = f2bf(e1 * hv.z + av.z);
    o.w = f2bf(e1 * hv.w + av.w);
    *(ushort4*)(zbf + i) = o;
}

// ---------------- fused MLP via bf16 WMMA: y = relu(z@W1+b1)@W2 + b2 ---------
// One wave handles a 16-node row tile. A fragments per CDNA5 ISA 16-bit A
// layout (lane<16: K = 0..7,16..23 ; lane>=16: K = 8..15,24..31, shifted by
// 32*s). B fragments read from transposed bf16 weights in the same K order:
// lane n<16 holds Wt[16c+n][32s+0..7, 32s+16..23], lane>=16 the +8 K-offset.
__global__ void __launch_bounds__(256)
k_mlp(const unsigned short* __restrict__ zbf,
      const unsigned short* __restrict__ w1t,
      const unsigned short* __restrict__ w2t,
      const float* __restrict__ b1, const float* __restrict__ b2,
      float* __restrict__ y) {
    __shared__ __align__(16) unsigned short tbuf[8][16 * DIM];

    int wave = threadIdx.x >> 5;
    int lane = threadIdx.x & 31;
    int tile = blockIdx.x * 8 + wave;
    if (tile >= NN / 16) return;

    int n0   = tile * 16;
    int mrow = lane & 15;
    int kh   = lane >> 4;

    // A fragments of z (4 k-steps of 32)
    Frag a[4];
    const unsigned short* zrow = zbf + (size_t)(n0 + mrow) * DIM;
    #pragma unroll
    for (int s = 0; s < 4; ++s) {
        int k0 = s * 32 + kh * 8;
        a[s].q[0] = *(const uint4*)(zrow + k0);
        a[s].q[1] = *(const uint4*)(zrow + k0 + 16);
    }

    // GEMM1: t = relu(z@W1 + b1) -> LDS (bf16, A-ready row-major)
    unsigned short* trow_base = tbuf[wave];
    #pragma unroll
    for (int c = 0; c < 8; ++c) {
        v8f acc = {0.f, 0.f, 0.f, 0.f, 0.f, 0.f, 0.f, 0.f};
        #pragma unroll
        for (int s = 0; s < 4; ++s) {
            Frag b;
            const unsigned short* wp =
                w1t + (size_t)(c * 16 + mrow) * DIM + s * 32 + kh * 8;
            b.q[0] = *(const uint4*)(wp);
            b.q[1] = *(const uint4*)(wp + 16);
            acc = __builtin_amdgcn_wmma_f32_16x16x32_bf16(
                false, a[s].v, false, b.v, (short)0, acc, false, false);
        }
        float bias = b1[c * 16 + mrow];
        #pragma unroll
        for (int r = 0; r < 8; ++r) {
            float v = acc[r] + bias;
            v = v > 0.f ? v : 0.f;
            int row = r + kh * 8;                      // C layout: M = r | 8+r
            trow_base[row * DIM + c * 16 + mrow] = f2bf(v);
        }
    }

    // A fragments of t from LDS (wave-private; DS ops in-order within wave)
    Frag a2[4];
    const unsigned short* trow = trow_base + (size_t)mrow * DIM;
    #pragma unroll
    for (int s = 0; s < 4; ++s) {
        int k0 = s * 32 + kh * 8;
        a2[s].q[0] = *(const uint4*)(trow + k0);
        a2[s].q[1] = *(const uint4*)(trow + k0 + 16);
    }

    // GEMM2: y = t@W2 + b2 -> global f32
    float* yout = y + (size_t)n0 * DIM;
    #pragma unroll
    for (int c = 0; c < 8; ++c) {
        v8f acc = {0.f, 0.f, 0.f, 0.f, 0.f, 0.f, 0.f, 0.f};
        #pragma unroll
        for (int s = 0; s < 4; ++s) {
            Frag b;
            const unsigned short* wp =
                w2t + (size_t)(c * 16 + mrow) * DIM + s * 32 + kh * 8;
            b.q[0] = *(const uint4*)(wp);
            b.q[1] = *(const uint4*)(wp + 16);
            acc = __builtin_amdgcn_wmma_f32_16x16x32_bf16(
                false, a2[s].v, false, b.v, (short)0, acc, false, false);
        }
        float bias = b2[c * 16 + mrow];
        #pragma unroll
        for (int r = 0; r < 8; ++r) {
            int row = r + kh * 8;
            yout[(size_t)row * DIM + c * 16 + mrow] = acc[r] + bias;
        }
    }
}

// -------- residual+mask in place over y, accumulate BN channel sums ----------
__global__ void k_res(const float* __restrict__ h, const float* __restrict__ mask,
                      float* __restrict__ y, float* __restrict__ bnsum,
                      float* __restrict__ bnsq) {
    int ch = threadIdx.x;                     // 0..127
    int nb = blockIdx.x * 64;
    float sum = 0.f, sq = 0.f;
    for (int r = 0; r < 64; ++r) {
        int node = nb + r;
        if (node >= NN) break;
        size_t idx = (size_t)node * DIM + ch;
        float v = mask[node] * y[idx] + h[idx];
        y[idx] = v;
        sum += v;
        sq  += v * v;
    }
    atomicAdd(&bnsum[ch], sum);
    atomicAdd(&bnsq[ch], sq);
}

// -------- BN finalize: scale/shift per channel -------------------------------
__global__ void k_bnfinal(const float* __restrict__ bnsum, const float* __restrict__ bnsq,
                          const float* __restrict__ gamma, const float* __restrict__ beta,
                          float* __restrict__ scale, float* __restrict__ shift) {
    int ch = threadIdx.x;
    float inv_n = 1.0f / (float)NN;
    float mu  = bnsum[ch] * inv_n;
    float var = bnsq[ch] * inv_n - mu * mu;
    float sc  = gamma[ch] * rsqrtf(var + BN_EPS);
    scale[ch] = sc;
    shift[ch] = beta[ch] - mu * sc;
}

// -------- BN apply + relu ----------------------------------------------------
__global__ void k_bnapply(const float* __restrict__ z2, const float* __restrict__ scale,
                          const float* __restrict__ shift, float* __restrict__ hout) {
    int i = (blockIdx.x * blockDim.x + threadIdx.x) * 4;
    if (i >= NN * DIM) return;
    int ch = i & (DIM - 1);
    float4 z = *(const float4*)(z2 + i);
    float4 o;
    o.x = fmaxf(scale[ch + 0] * z.x + shift[ch + 0], 0.f);
    o.y = fmaxf(scale[ch + 1] * z.y + shift[ch + 1], 0.f);
    o.z = fmaxf(scale[ch + 2] * z.z + shift[ch + 2], 0.f);
    o.w = fmaxf(scale[ch + 3] * z.w + shift[ch + 3], 0.f);
    *(float4*)(hout + i) = o;
}

extern "C" void kernel_launch(void* const* d_in, const int* in_sizes, int n_in,
                              void* d_out, int out_size, void* d_ws, size_t ws_size,
                              hipStream_t stream) {
    const float* x         = (const float*)d_in[0];
    const float* edge_attr = (const float*)d_in[1];
    const float* masks     = (const float*)d_in[2];
    const int*   edge_idx  = (const int*)d_in[3];
    const float* eps       = (const float*)d_in[4];
    const float* W1        = (const float*)d_in[5];
    const float* b1        = (const float*)d_in[6];
    const float* W2        = (const float*)d_in[7];
    const float* b2        = (const float*)d_in[8];
    const float* gamma     = (const float*)d_in[9];
    const float* beta      = (const float*)d_in[10];
    float* out = (float*)d_out;

    // workspace carve-up (256B aligned)
    char* ws = (char*)d_ws;
    size_t off = 0;
    auto carve = [&](size_t bytes) -> char* {
        char* p = ws + off;
        off += (bytes + 255) & ~(size_t)255;
        return p;
    };
    float*          agg   = (float*)         carve((size_t)NN * DIM * 4);
    unsigned short* zbf   = (unsigned short*)carve((size_t)NN * DIM * 2);
    float*          ybuf  = (float*)         carve((size_t)NN * DIM * 4); // y, then z2
    float*          hbuf  = (float*)         carve((size_t)NN * DIM * 4);
    unsigned short* wt    = (unsigned short*)carve((size_t)LAYERS * 2 * DIM * DIM * 2);
    float*          bnsum = (float*)         carve(DIM * 4);
    float*          bnsq  = (float*)         carve(DIM * 4);
    float*          bnsc  = (float*)         carve(DIM * 4);
    float*          bnsh  = (float*)         carve(DIM * 4);
    (void)ws_size; (void)in_sizes; (void)n_in; (void)out_size;

    k_wprep<<<(LAYERS * 2 * DIM * DIM + 255) / 256, 256, 0, stream>>>(W1, W2, wt);

    for (int l = 0; l < LAYERS; ++l) {
        const float* hcur = (l == 0) ? x : hbuf;
        float* hnext = (l == LAYERS - 1) ? out : hbuf;

        hipMemsetAsync(agg, 0, (size_t)NN * DIM * 4, stream);
        hipMemsetAsync(bnsum, 0, DIM * 4, stream);
        hipMemsetAsync(bnsq, 0, DIM * 4, stream);

        k_scatter<<<NE / 8, 256, 0, stream>>>(hcur, edge_attr, edge_idx, agg);
        k_zprep<<<(NN * DIM / 4 + 255) / 256, 256, 0, stream>>>(hcur, agg, eps, l, zbf);
        k_mlp<<<(NN / 16 + 7) / 8, 256, 0, stream>>>(
            zbf, wt + (size_t)(l * 2 + 0) * DIM * DIM, wt + (size_t)(l * 2 + 1) * DIM * DIM,
            b1 + (size_t)l * DIM, b2 + (size_t)l * DIM, ybuf);
        k_res<<<(NN + 63) / 64, 128, 0, stream>>>(hcur, masks + (size_t)l * NN, ybuf,
                                                  bnsum, bnsq);
        k_bnfinal<<<1, 128, 0, stream>>>(bnsum, bnsq, gamma + (size_t)l * DIM,
                                         beta + (size_t)l * DIM, bnsc, bnsh);
        k_bnapply<<<(NN * DIM / 4 + 255) / 256, 256, 0, stream>>>(ybuf, bnsc, bnsh, hnext);
    }
}
